// TemporallyCalibratedTransformerLayer_42039139893866
// MI455X (gfx1250) — compile-verified, matrix operand order checked by reading, not run on hardware
//
#include <hip/hip_runtime.h>
#include <hip/hip_bf16.h>
#include <math.h>

// ---------------------------------------------------------------------------
// TemporallyCalibratedTransformerLayer on gfx1250 (MI455X), bf16 WMMA path.
// Round 2: 32x64 GEMM tiles + software-pipelined fragment loads so WMMAs
// overlap with in-flight global_load_b128 (avoid s_wait_loadcnt 0).
// ---------------------------------------------------------------------------

typedef __bf16 bf16;
typedef __attribute__((ext_vector_type(16))) __bf16 v16bf;
typedef __attribute__((ext_vector_type(8)))  float  v8f;
typedef __attribute__((ext_vector_type(4)))  float  f32x4;

#define DIM   768
#define FDIM  3072
#define NTOK  197
#define BATCH 64
#define MROWS (BATCH * NTOK)   /* 12608 = 394*32 */
#define HEADS 12
#define DH    64
#define NPAD  224              /* 197 padded to 14*16, multiple of 32 */
#define TM32  (MROWS / 32)     /* 394 M-tiles of 32 rows */

// ---------------- WMMA helpers -------------------------------------------

__device__ __forceinline__ v8f wmma_bf16(v16bf a, v16bf b, v8f c) {
  return __builtin_amdgcn_wmma_f32_16x16x32_bf16(false, a, false, b,
                                                 (short)0, c, false, false);
}

// Load a 16x32 bf16 fragment (A operand, or B operand via its transposed
// source matrix) from a row-major matrix with leading dim `ld` (elements).
// Per ISA 7.12.2: lanes 0-15 hold row (row0+lane), K chunks {k0..k0+7,
// k0+16..k0+23}; lanes 16-31 hold the same rows, K chunks {+8, +24}.
__device__ __forceinline__ v16bf load_frag_g(const bf16* __restrict__ base,
                                             int ld, int row0, int k0, int lane) {
  const int r  = row0 + (lane & 15);
  const int kk = k0 + ((lane >> 4) << 3);
  const bf16* p = base + (size_t)r * ld + kk;
  union { v16bf v; struct { f32x4 lo, hi; } q; } u;
  u.q.lo = *reinterpret_cast<const f32x4*>(p);
  u.q.hi = *reinterpret_cast<const f32x4*>(p + 16);
  return u.v;
}

// Same fragment pattern, but source rows are fp32 in LDS (converted to bf16).
__device__ __forceinline__ v16bf load_frag_lds_f32(const float* base, int ld,
                                                   int k0, int lane) {
  const int r  = lane & 15;
  const int kk = k0 + ((lane >> 4) << 3);
  const float* p = base + r * ld + kk;
  v16bf f;
#pragma unroll
  for (int i = 0; i < 8; ++i) {
    f[i]     = (bf16)p[i];
    f[i + 8] = (bf16)p[i + 16];
  }
  return f;
}

// One wave computes a 32(M) x 64(N) tile: 8 accumulators, A/B fragment reuse,
// one-step software pipeline (prefetch k+32 while issuing WMMAs for k).
__device__ __forceinline__ void gemm_tile32(const bf16* __restrict__ A,
                                            const bf16* __restrict__ W,
                                            int K, int row0, int col0, int lane,
                                            v8f acc[8]) {
#pragma unroll
  for (int j = 0; j < 8; ++j)
#pragma unroll
    for (int i = 0; i < 8; ++i) acc[j][i] = 0.f;

  v16bf a0 = load_frag_g(A, K, row0,      0, lane);
  v16bf a1 = load_frag_g(A, K, row0 + 16, 0, lane);
  v16bf b0 = load_frag_g(W, K, col0,      0, lane);
  v16bf b1 = load_frag_g(W, K, col0 + 16, 0, lane);
  v16bf b2 = load_frag_g(W, K, col0 + 32, 0, lane);
  v16bf b3 = load_frag_g(W, K, col0 + 48, 0, lane);

  for (int k0 = 32; k0 < K; k0 += 32) {
    v16bf na0 = load_frag_g(A, K, row0,      k0, lane);
    v16bf na1 = load_frag_g(A, K, row0 + 16, k0, lane);
    v16bf nb0 = load_frag_g(W, K, col0,      k0, lane);
    v16bf nb1 = load_frag_g(W, K, col0 + 16, k0, lane);
    v16bf nb2 = load_frag_g(W, K, col0 + 32, k0, lane);
    v16bf nb3 = load_frag_g(W, K, col0 + 48, k0, lane);
    acc[0] = wmma_bf16(a0, b0, acc[0]);
    acc[1] = wmma_bf16(a0, b1, acc[1]);
    acc[2] = wmma_bf16(a0, b2, acc[2]);
    acc[3] = wmma_bf16(a0, b3, acc[3]);
    acc[4] = wmma_bf16(a1, b0, acc[4]);
    acc[5] = wmma_bf16(a1, b1, acc[5]);
    acc[6] = wmma_bf16(a1, b2, acc[6]);
    acc[7] = wmma_bf16(a1, b3, acc[7]);
    a0 = na0; a1 = na1; b0 = nb0; b1 = nb1; b2 = nb2; b3 = nb3;
  }
  acc[0] = wmma_bf16(a0, b0, acc[0]);
  acc[1] = wmma_bf16(a0, b1, acc[1]);
  acc[2] = wmma_bf16(a0, b2, acc[2]);
  acc[3] = wmma_bf16(a0, b3, acc[3]);
  acc[4] = wmma_bf16(a1, b0, acc[4]);
  acc[5] = wmma_bf16(a1, b1, acc[5]);
  acc[6] = wmma_bf16(a1, b2, acc[6]);
  acc[7] = wmma_bf16(a1, b3, acc[7]);
}

// ---------------- small utility kernels ----------------------------------

__global__ void cvt_bf16_kernel(const float* __restrict__ in,
                                bf16* __restrict__ out, int n) {
  int i = blockIdx.x * 256 + threadIdx.x;
  int stride = gridDim.x * 256;
  for (; i < n; i += stride) out[i] = (bf16)in[i];
}

__global__ void zero_kernel(unsigned int* __restrict__ p, long long n) {
  long long i = (long long)blockIdx.x * blockDim.x + threadIdx.x;
  long long stride = (long long)gridDim.x * blockDim.x;
  for (; i < n; i += stride) p[i] = 0u;
}

// LayerNorm over last dim (768), output bf16. One block (256 thr) per row.
__global__ __launch_bounds__(256)
void ln_kernel(const float* __restrict__ x, const float* __restrict__ w,
               const float* __restrict__ b, bf16* __restrict__ out) {
  const int row = blockIdx.x;
  const int tid = threadIdx.x;
  const float* xr = x + (size_t)row * DIM;
  float s = 0.f, ss = 0.f;
  for (int i = tid; i < DIM; i += 256) {
    float v = xr[i];
    s += v; ss += v * v;
  }
  __shared__ float red0[256];
  __shared__ float red1[256];
  red0[tid] = s; red1[tid] = ss;
  __syncthreads();
  for (int off = 128; off > 0; off >>= 1) {
    if (tid < off) { red0[tid] += red0[tid + off]; red1[tid] += red1[tid + off]; }
    __syncthreads();
  }
  const float mean = red0[0] * (1.f / DIM);
  const float var  = red1[0] * (1.f / DIM) - mean * mean;
  const float inv  = rsqrtf(var + 1e-6f);
  for (int i = tid; i < DIM; i += 256) {
    float v = (xr[i] - mean) * inv * w[i] + b[i];
    out[(size_t)row * DIM + i] = (bf16)v;
  }
}

// Temporal conv1d (k=3, pad=1) over per-frame CLS tokens; one thread per
// (which, b_global, cout).  g[bv][cin][t] = h[(bv*4+t)*197 + 0][cin].
__global__ void conv_ctx_kernel(const bf16* __restrict__ h,
                                const float* __restrict__ wq, const float* __restrict__ bq,
                                const float* __restrict__ wk, const float* __restrict__ bk,
                                float* __restrict__ cq, float* __restrict__ ck) {
  int idx = blockIdx.x * blockDim.x + threadIdx.x;
  if (idx >= 2 * BATCH * DIM) return;
  const int which = idx / (BATCH * DIM);
  const int rem   = idx % (BATCH * DIM);
  const int bg    = rem / DIM;     // global batch = bv*4 + t
  const int cout  = rem % DIM;
  const int bv = bg >> 2, t = bg & 3;
  const float* w    = which ? wk : wq;
  const float* bias = which ? bk : bq;
  float sum = bias[cout];
  for (int dt = 0; dt < 3; ++dt) {
    const int tt = t + dt - 1;
    if (tt < 0 || tt >= 4) continue;
    const bf16* g   = h + (size_t)((bv * 4 + tt) * NTOK) * DIM;  // CLS row
    const float* wr = w + (size_t)cout * DIM * 3 + dt;           // (Cout,Cin,3)
    for (int cin = 0; cin < DIM; ++cin)
      sum += wr[cin * 3] * (float)g[cin];
  }
  (which ? ck : cq)[rem] = sum;
}

// ---------------- GEMM kernels -------------------------------------------

// h @ qkv_w^T + qkv_b, plus temporal context; scatter to padded per-head
// qa/ka (NPAD x 64, row-major) and vT (64 x NPAD).
__global__ __launch_bounds__(256)
void qkv_gemm_kernel(const bf16* __restrict__ A, const bf16* __restrict__ W,
                     const float* __restrict__ bias,
                     const float* __restrict__ cq, const float* __restrict__ ck,
                     bf16* __restrict__ qa, bf16* __restrict__ ka,
                     bf16* __restrict__ vt) {
  const int lane = threadIdx.x & 31;
  const int wave = threadIdx.x >> 5;
  const int tilesN = (3 * DIM) / 64;  // 36
  const int fw = blockIdx.x * 8 + wave;
  const int tn = fw % tilesN, tm = fw / tilesN;
  if (tm >= TM32) return;
  const int row0 = tm * 32, col0 = tn * 64;
  v8f acc[8];
  gemm_tile32(A, W, DIM, row0, col0, lane, acc);

  const int mh = (lane >> 4) << 3;
  const int nl = lane & 15;
#pragma unroll
  for (int half = 0; half < 2; ++half) {
#pragma unroll
    for (int j = 0; j < 4; ++j) {
      const int col = col0 + j * 16 + nl;
      const int s   = col / DIM;        // uniform per 64-col block
      const int rem = col % DIM;
      const int hh  = rem / DH, d = rem % DH;
#pragma unroll
      for (int r = 0; r < 8; ++r) {
        const int m = row0 + half * 16 + r + mh;
        const int b = m / NTOK, n = m % NTOK;
        float val = acc[half * 4 + j][r] + bias[col];
        if (s == 0) {
          val += cq[b * DIM + rem];
          qa[((size_t)(b * HEADS + hh) * NPAD + n) * DH + d] = (bf16)val;
        } else if (s == 1) {
          val += ck[b * DIM + rem];
          ka[((size_t)(b * HEADS + hh) * NPAD + n) * DH + d] = (bf16)val;
        } else {
          vt[((size_t)(b * HEADS + hh) * DH + d) * NPAD + n] = (bf16)val;
        }
      }
    }
  }
}

// Per-head attention: scores (WMMA) -> softmax in LDS -> attn@V (WMMA).
// 4 waves per block, one block per (b,h); each wave owns private 16xNPAD band.
__global__ __launch_bounds__(128)
void attn_kernel(const bf16* __restrict__ qa, const bf16* __restrict__ ka,
                 const bf16* __restrict__ vt, bf16* __restrict__ O) {
  __shared__ float sS[4 * 16 * NPAD];  // 56 KB
  const int lane = threadIdx.x & 31;
  const int wave = threadIdx.x >> 5;
  const int head = blockIdx.x;
  const int b = head / HEADS, hh = head % HEADS;
  const bf16* qh = qa + (size_t)head * NPAD * DH;
  const bf16* kh = ka + (size_t)head * NPAD * DH;
  const bf16* vh = vt + (size_t)head * DH * NPAD;
  float* Sb = sS + wave * 16 * NPAD;
  const int mh = (lane >> 4) << 3;
  const int nl = lane & 15;

  for (int band = wave; band < NPAD / 16; band += 4) {
    const int row0 = band * 16;
    // ---- S band = scale * (q+cq) @ (k+ck)^T ----  (pipelined K fragments)
    v16bf a0 = load_frag_g(qh, DH, row0, 0, lane);
    v16bf a1 = load_frag_g(qh, DH, row0, 32, lane);
    v8f acc[NPAD / 16];
    v16bf kb0 = load_frag_g(kh, DH, 0, 0, lane);
    v16bf kb1 = load_frag_g(kh, DH, 0, 32, lane);
#pragma unroll
    for (int j = 0; j < NPAD / 16; ++j) {
      const int jn = (j + 1 < NPAD / 16) ? j + 1 : j;
      v16bf nb0 = load_frag_g(kh, DH, jn * 16, 0, lane);
      v16bf nb1 = load_frag_g(kh, DH, jn * 16, 32, lane);
#pragma unroll
      for (int i = 0; i < 8; ++i) acc[j][i] = 0.f;
      acc[j] = wmma_bf16(a0, kb0, acc[j]);
      acc[j] = wmma_bf16(a1, kb1, acc[j]);
      kb0 = nb0; kb1 = nb1;
    }
#pragma unroll
    for (int j = 0; j < NPAD / 16; ++j)
#pragma unroll
      for (int r = 0; r < 8; ++r)
        Sb[(r + mh) * NPAD + j * 16 + nl] = acc[j][r] * 0.125f;

    // ---- masked softmax, one row per lane (lanes 0..15) ----
    if (lane < 16) {
      float* rp = Sb + lane * NPAD;
      float mx = -1e30f;
      for (int c = 0; c < NTOK; ++c) mx = fmaxf(mx, rp[c]);
      float sum = 0.f;
      for (int c = 0; c < NTOK; ++c) { float e = __expf(rp[c] - mx); rp[c] = e; sum += e; }
      const float inv = 1.f / sum;
      for (int c = 0; c < NTOK; ++c) rp[c] *= inv;
      for (int c = NTOK; c < NPAD; ++c) rp[c] = 0.f;
    }

    // ---- o band = attn @ v ----  (pipelined V fragments)
#pragma unroll
    for (int t = 0; t < DH / 16; ++t) {
      v8f oa;
#pragma unroll
      for (int i = 0; i < 8; ++i) oa[i] = 0.f;
      v16bf bcur = load_frag_g(vh, NPAD, t * 16, 0, lane);
      for (int k0 = 0; k0 < NPAD; k0 += 32) {
        const int kn = (k0 + 32 < NPAD) ? k0 + 32 : k0;
        v16bf bnext = load_frag_g(vh, NPAD, t * 16, kn, lane);
        v16bf af = load_frag_lds_f32(Sb, NPAD, k0, lane);
        oa = wmma_bf16(af, bcur, oa);
        bcur = bnext;
      }
#pragma unroll
      for (int r = 0; r < 8; ++r) {
        const int m = row0 + r + mh;
        if (m < NTOK)
          O[((size_t)(b * NTOK + m)) * DIM + hh * DH + t * 16 + nl] = (bf16)oa[r];
      }
    }
  }
}

// x1 = x + O @ proj_w^T + proj_b   (fp32 out -> d_out)
__global__ __launch_bounds__(256)
void proj_gemm_kernel(const bf16* __restrict__ A, const bf16* __restrict__ W,
                      const float* __restrict__ bias,
                      const float* __restrict__ resid, float* __restrict__ out) {
  const int lane = threadIdx.x & 31;
  const int wave = threadIdx.x >> 5;
  const int tilesN = DIM / 64;  // 12
  const int fw = blockIdx.x * 8 + wave;
  const int tn = fw % tilesN, tm = fw / tilesN;
  if (tm >= TM32) return;
  const int row0 = tm * 32, col0 = tn * 64;
  v8f acc[8];
  gemm_tile32(A, W, DIM, row0, col0, lane, acc);
  const int mh = (lane >> 4) << 3, nl = lane & 15;
#pragma unroll
  for (int half = 0; half < 2; ++half)
#pragma unroll
    for (int j = 0; j < 4; ++j) {
      const int col = col0 + j * 16 + nl;
#pragma unroll
      for (int r = 0; r < 8; ++r) {
        const size_t idx = (size_t)(row0 + half * 16 + r + mh) * DIM + col;
        out[idx] = acc[half * 4 + j][r] + bias[col] + resid[idx];
      }
    }
}

// g1 = gelu_exact(h2 @ ff1_w^T + ff1_b)  (bf16 out)
__global__ __launch_bounds__(256)
void ff1_gemm_kernel(const bf16* __restrict__ A, const bf16* __restrict__ W,
                     const float* __restrict__ bias, bf16* __restrict__ out) {
  const int lane = threadIdx.x & 31;
  const int wave = threadIdx.x >> 5;
  const int tilesN = FDIM / 64;  // 48
  const int fw = blockIdx.x * 8 + wave;
  const int tn = fw % tilesN, tm = fw / tilesN;
  if (tm >= TM32) return;
  const int row0 = tm * 32, col0 = tn * 64;
  v8f acc[8];
  gemm_tile32(A, W, DIM, row0, col0, lane, acc);
  const int mh = (lane >> 4) << 3, nl = lane & 15;
#pragma unroll
  for (int half = 0; half < 2; ++half)
#pragma unroll
    for (int j = 0; j < 4; ++j) {
      const int col = col0 + j * 16 + nl;
#pragma unroll
      for (int r = 0; r < 8; ++r) {
        float v = acc[half * 4 + j][r] + bias[col];
        v = 0.5f * v * (1.f + erff(v * 0.70710678118654752f));  // exact gelu
        out[(size_t)(row0 + half * 16 + r + mh) * FDIM + col] = (bf16)v;
      }
    }
}

// out += g1 @ ff2_w^T + ff2_b   (accumulate into d_out which holds x1)
__global__ __launch_bounds__(256)
void ff2_gemm_kernel(const bf16* __restrict__ A, const bf16* __restrict__ W,
                     const float* __restrict__ bias, float* __restrict__ out) {
  const int lane = threadIdx.x & 31;
  const int wave = threadIdx.x >> 5;
  const int tilesN = DIM / 64;  // 12
  const int fw = blockIdx.x * 8 + wave;
  const int tn = fw % tilesN, tm = fw / tilesN;
  if (tm >= TM32) return;
  const int row0 = tm * 32, col0 = tn * 64;
  v8f acc[8];
  gemm_tile32(A, W, FDIM, row0, col0, lane, acc);
  const int mh = (lane >> 4) << 3, nl = lane & 15;
#pragma unroll
  for (int half = 0; half < 2; ++half)
#pragma unroll
    for (int j = 0; j < 4; ++j) {
      const int col = col0 + j * 16 + nl;
#pragma unroll
      for (int r = 0; r < 8; ++r) {
        const size_t idx = (size_t)(row0 + half * 16 + r + mh) * DIM + col;
        out[idx] += acc[half * 4 + j][r] + bias[col];
      }
    }
}

// ---------------- launch --------------------------------------------------

extern "C" void kernel_launch(void* const* d_in, const int* in_sizes, int n_in,
                              void* d_out, int out_size, void* d_ws, size_t ws_size,
                              hipStream_t stream) {
  const float* x      = (const float*)d_in[0];
  const float* ln1_w  = (const float*)d_in[1];
  const float* ln1_b  = (const float*)d_in[2];
  const float* qkv_w  = (const float*)d_in[3];
  const float* qkv_b  = (const float*)d_in[4];
  const float* convq_w= (const float*)d_in[5];
  const float* convq_b= (const float*)d_in[6];
  const float* convk_w= (const float*)d_in[7];
  const float* convk_b= (const float*)d_in[8];
  const float* proj_w = (const float*)d_in[9];
  const float* proj_b = (const float*)d_in[10];
  const float* ln2_w  = (const float*)d_in[11];
  const float* ln2_b  = (const float*)d_in[12];
  const float* ff1_w  = (const float*)d_in[13];
  const float* ff1_b  = (const float*)d_in[14];
  const float* ff2_w  = (const float*)d_in[15];
  const float* ff2_b  = (const float*)d_in[16];
  float* out = (float*)d_out;

  char* ws = (char*)d_ws;
  size_t off = 0;
  auto alloc = [&](size_t bytes) -> char* {
    char* p = ws + off;
    off += (bytes + 255) & ~(size_t)255;
    return p;
  };
  bf16*  wqkv  = (bf16*)alloc((size_t)3 * DIM * DIM * 2);
  bf16*  wproj = (bf16*)alloc((size_t)DIM * DIM * 2);
  bf16*  wff1  = (bf16*)alloc((size_t)FDIM * DIM * 2);
  bf16*  wff2  = (bf16*)alloc((size_t)DIM * FDIM * 2);
  float* cqb   = (float*)alloc((size_t)BATCH * DIM * 4);
  float* ckb   = (float*)alloc((size_t)BATCH * DIM * 4);
  bf16*  hb    = (bf16*)alloc((size_t)MROWS * DIM * 2);     // LN1 out; reused for LN2
  char*  qbase = ws + off;
  const size_t headBytes = (size_t)BATCH * HEADS * NPAD * DH * 2;
  bf16*  qa    = (bf16*)alloc(headBytes);
  bf16*  ka    = (bf16*)alloc(headBytes);
  bf16*  vt    = (bf16*)alloc(headBytes);
  bf16*  Ob    = (bf16*)alloc((size_t)MROWS * DIM * 2);
  // FF1 intermediate (77.4MB) aliases the freed qa..Ob span (85.4MB):
  bf16*  g1    = (bf16*)qbase;

  // 1. weights -> bf16
  cvt_bf16_kernel<<<6912, 256, 0, stream>>>(qkv_w, wqkv, 3 * DIM * DIM);
  cvt_bf16_kernel<<<2304, 256, 0, stream>>>(proj_w, wproj, DIM * DIM);
  cvt_bf16_kernel<<<9216, 256, 0, stream>>>(ff1_w, wff1, FDIM * DIM);
  cvt_bf16_kernel<<<9216, 256, 0, stream>>>(ff2_w, wff2, DIM * FDIM);
  // 2. LN1
  ln_kernel<<<MROWS, 256, 0, stream>>>(x, ln1_w, ln1_b, hb);
  // 3. temporal context convs
  conv_ctx_kernel<<<(2 * BATCH * DIM + 255) / 256, 256, 0, stream>>>(
      hb, convq_w, convq_b, convk_w, convk_b, cqb, ckb);
  // 4. zero padded qa/ka/vt region
  zero_kernel<<<8192, 256, 0, stream>>>((unsigned int*)qa,
                                        (long long)(3 * headBytes / 4));
  // 5. QKV GEMM + context add + scatter  (394*36/8 = 1773 blocks)
  qkv_gemm_kernel<<<TM32 * 36 / 8, 256, 0, stream>>>(
      hb, wqkv, qkv_b, cqb, ckb, qa, ka, vt);
  // 6. attention per head
  attn_kernel<<<BATCH * HEADS, 128, 0, stream>>>(qa, ka, vt, Ob);
  // 7. proj + residual -> d_out (holds x1)   (394*12/8 = 591 blocks)
  proj_gemm_kernel<<<TM32 * 12 / 8, 256, 0, stream>>>(
      Ob, wproj, proj_b, x, out);
  // 8. LN2
  ln_kernel<<<MROWS, 256, 0, stream>>>(out, ln2_w, ln2_b, hb);
  // 9. FF1 + gelu   (394*48/8 = 2364 blocks)
  ff1_gemm_kernel<<<TM32 * 48 / 8, 256, 0, stream>>>(hb, wff1, ff1_b, g1);
  // 10. FF2 accumulate into d_out
  ff2_gemm_kernel<<<TM32 * 12 / 8, 256, 0, stream>>>(g1, wff2, ff2_b, out);
}
